// L4QQuantizedLinear_30837865185896
// MI455X (gfx1250) — matrix operand order repaired and checked by reading, not compiled, not run on hardware
//
#include <hip/hip_runtime.h>
#include <hip/hip_bf16.h>

typedef __attribute__((ext_vector_type(16))) _Float16 v16h;
typedef __attribute__((ext_vector_type(8)))  float    v8f;
typedef __attribute__((ext_vector_type(4)))  float    v4f;
typedef __attribute__((ext_vector_type(4)))  _Float16 v4h;

#define IN_F   4096
#define OUT_F  4096
#define RANK   16
#define QEPS   1e-9f

// ------------------------------------------------------------------
// Kernel 1: w_q = quant(w0 + lora_b@lora_a), stored as f16 hi/lo planes
// grid (IN_F/256, OUT_F/64), block 256
// ------------------------------------------------------------------
#define BO 64
#define BI 256

__global__ __launch_bounds__(256) void wq_split_kernel(
    const float* __restrict__ w0, const float* __restrict__ lora_a,
    const float* __restrict__ lora_b, const float* __restrict__ q_scale,
    _Float16* __restrict__ whi, _Float16* __restrict__ wlo)
{
    __shared__ float sA[RANK][BI];   // lora_a[:, i0:i0+256]
    __shared__ float sB[BO][RANK];   // lora_b[o0:o0+64, :]

    const int t  = threadIdx.x;
    const int i0 = blockIdx.x * BI;
    const int o0 = blockIdx.y * BO;

    #pragma unroll
    for (int j = 0; j < 4; ++j) {
        int f4  = t + 256 * j;          // float4 id 0..1023
        int row = f4 >> 6;              // 64 float4 per row
        int col = (f4 & 63) << 2;
        *(v4f*)&sA[row][col] = *(const v4f*)&lora_a[(size_t)row * IN_F + i0 + col];
    }
    {
        int row = t >> 2;
        int r0  = (t & 3) << 2;
        *(v4f*)&sB[row][r0] = *(const v4f*)&lora_b[(size_t)(o0 + row) * RANK + r0];
    }
    __syncthreads();

    const int row    = t >> 2;
    const int o      = o0 + row;
    const int colgrp = t & 3;           // 4 threads per row, 64 cols each

    float b[RANK];
    #pragma unroll
    for (int r = 0; r < RANK; ++r) b[r] = sB[row][r];

    #pragma unroll 4
    for (int c4 = 0; c4 < 16; ++c4) {
        int    il  = colgrp * 64 + c4 * 4;
        size_t idx = (size_t)o * IN_F + i0 + il;
        v4f w = *(const v4f*)&w0[idx];
        #pragma unroll
        for (int r = 0; r < RANK; ++r) {
            float br = b[r];
            w.x += br * sA[r][il + 0];
            w.y += br * sA[r][il + 1];
            w.z += br * sA[r][il + 2];
            w.w += br * sA[r][il + 3];
        }
        float s  = q_scale[idx >> 7];   // GROUP=128, 4 elems share a group
        float sd = s + QEPS;
        float wv[4] = { w.x, w.y, w.z, w.w };
        v4h hi4, lo4;
        #pragma unroll
        for (int j = 0; j < 4; ++j) {
            float ws = wv[j] / sd;
            ws = fminf(fmaxf(ws, -8.0f), 7.0f);
            float wq = rintf(ws) * s;        // RNE matches jnp.round
            _Float16 h = (_Float16)wq;
            hi4[j] = h;
            lo4[j] = (_Float16)(wq - (float)h);
        }
        *(v4h*)&whi[idx] = hi4;
        *(v4h*)&wlo[idx] = lo4;
    }
}

// ------------------------------------------------------------------
// Kernel 1b: split x (f32) into f16 hi/lo planes, one pass, memory-bound
// ------------------------------------------------------------------
__global__ __launch_bounds__(256) void x_split_kernel(
    const float* __restrict__ x, _Float16* __restrict__ xhi,
    _Float16* __restrict__ xlo)
{
    size_t base = ((size_t)blockIdx.x * 256 + threadIdx.x) * 4;
    v4f v = *(const v4f*)&x[base];
    float vv[4] = { v.x, v.y, v.z, v.w };
    v4h hi4, lo4;
    #pragma unroll
    for (int j = 0; j < 4; ++j) {
        _Float16 h = (_Float16)vv[j];
        hi4[j] = h;
        lo4[j] = (_Float16)(vv[j] - (float)h);
    }
    *(v4h*)&xhi[base] = hi4;
    *(v4h*)&xlo[base] = lo4;
}

// ------------------------------------------------------------------
// Kernel 2: out = x @ w_q^T + bias via split-f16 WMMA (hi*hi+hi*lo+lo*hi)
// Block tile 128x256, 8 wave32 waves in 2x4 grid, wave tile 64x64.
// Per k-step per wave: 32 ds_load_b128 feed 48 v_wmma (1.5 wmma/load).
// ------------------------------------------------------------------
#define BM 128
#define BN 256
#define BK 32
#define WP 40   // padded f16 LDS row: 80B stride -> conflict-free b128 reads

__global__ __launch_bounds__(256) void gemm_wmma_kernel(
    const _Float16* __restrict__ xhi, const _Float16* __restrict__ xlo,
    const _Float16* __restrict__ whi, const _Float16* __restrict__ wlo,
    const float* __restrict__ bias, float* __restrict__ out)
{
    __shared__ _Float16 sXH[BM * WP];     // 10 KB
    __shared__ _Float16 sXL[BM * WP];     // 10 KB
    __shared__ _Float16 sWH[BN * WP];     // 20 KB
    __shared__ _Float16 sWL[BN * WP];     // 20 KB

    const int t    = threadIdx.x;
    const int lane = t & 31;
    const int wv   = t >> 5;       // wave 0..7
    const int wm   = wv & 1;       // 2 waves along M (64 rows each)
    const int wn   = wv >> 1;      // 4 waves along N (64 cols each)
    const int m0   = blockIdx.y * BM;
    const int n0   = blockIdx.x * BN;

    // global->LDS tile mapping: 2 threads per row, 16 halves per thread.
    // x planes: rows 0..127.  w planes: rows (t>>1) and 128+(t>>1).
    const int lrow = t >> 1;            // 0..127
    const int lcol = (t & 1) << 4;      // 0 or 16 (halves)

    const _Float16* gxh = xhi + (size_t)(m0 + lrow) * IN_F + lcol;
    const _Float16* gxl = xlo + (size_t)(m0 + lrow) * IN_F + lcol;
    const _Float16* gwh = whi + (size_t)(n0 + lrow) * IN_F + lcol;
    const _Float16* gwl = wlo + (size_t)(n0 + lrow) * IN_F + lcol;
    const size_t    wstep = (size_t)128 * IN_F;     // second w row block

    v8f acc[4][4];
    #pragma unroll
    for (int mi = 0; mi < 4; ++mi)
        #pragma unroll
        for (int ni = 0; ni < 4; ++ni)
            acc[mi][ni] = (v8f){0.f,0.f,0.f,0.f,0.f,0.f,0.f,0.f};

    // register prefetch buffers (overlap HBM latency with WMMA)
    v4f rxh[2], rxl[2], rwh[4], rwl[4];
    #pragma unroll
    for (int j = 0; j < 2; ++j) {
        rxh[j]     = *(const v4f*)(gxh + 8 * j);           // 8 f16 = 16 B
        rxl[j]     = *(const v4f*)(gxl + 8 * j);
        rwh[j]     = *(const v4f*)(gwh + 8 * j);
        rwl[j]     = *(const v4f*)(gwl + 8 * j);
        rwh[2 + j] = *(const v4f*)(gwh + wstep + 8 * j);
        rwl[2 + j] = *(const v4f*)(gwl + wstep + 8 * j);
    }

    // per-lane fragment offsets (ISA VGPR layouts, wave32):
    const int kA  = (lane >> 4) << 3;   // A: lanes16-31 hold K {8-15, 24-31}
    const int kB  = (lane >> 4) << 3;   // B: lanes16-31 hold K {8-15, 24-31}
    const int l16 = lane & 15;

    for (int k0 = 0; k0 < IN_F; k0 += BK) {
        // stage prefetched tile into LDS
        #pragma unroll
        for (int j = 0; j < 2; ++j) {
            *(v4f*)&sXH[lrow * WP + lcol + 8 * j]         = rxh[j];
            *(v4f*)&sXL[lrow * WP + lcol + 8 * j]         = rxl[j];
            *(v4f*)&sWH[lrow * WP + lcol + 8 * j]         = rwh[j];
            *(v4f*)&sWL[lrow * WP + lcol + 8 * j]         = rwl[j];
            *(v4f*)&sWH[(128 + lrow) * WP + lcol + 8 * j] = rwh[2 + j];
            *(v4f*)&sWL[(128 + lrow) * WP + lcol + 8 * j] = rwl[2 + j];
        }
        __syncthreads();

        if (k0 + BK < IN_F) {
            gxh += BK; gxl += BK; gwh += BK; gwl += BK;
            #pragma unroll
            for (int j = 0; j < 2; ++j) {
                rxh[j]     = *(const v4f*)(gxh + 8 * j);
                rxl[j]     = *(const v4f*)(gxl + 8 * j);
                rwh[j]     = *(const v4f*)(gwh + 8 * j);
                rwl[j]     = *(const v4f*)(gwl + 8 * j);
                rwh[2 + j] = *(const v4f*)(gwh + wstep + 8 * j);
                rwl[2 + j] = *(const v4f*)(gwl + wstep + 8 * j);
            }
        }

        // B fragments: lane = column n, K halves {kB..kB+7, kB+16..kB+23}
        v16h bh[4], bl[4];
        #pragma unroll
        for (int ni = 0; ni < 4; ++ni) {
            int nrow = wn * 64 + ni * 16 + l16;
            const _Float16* ph = &sWH[nrow * WP + kB];
            const _Float16* pl = &sWL[nrow * WP + kB];
            union { v16h h; v4f f[2]; } uh, ul;
            uh.f[0] = *(const v4f*)ph;  uh.f[1] = *(const v4f*)(ph + 16);
            ul.f[0] = *(const v4f*)pl;  ul.f[1] = *(const v4f*)(pl + 16);
            bh[ni] = uh.h;  bl[ni] = ul.h;
        }

        // A fragments: lane = row m, K halves {kA..kA+7, kA+16..kA+23}
        #pragma unroll
        for (int mi = 0; mi < 4; ++mi) {
            int mrow = wm * 64 + mi * 16 + l16;
            const _Float16* ph = &sXH[mrow * WP + kA];
            const _Float16* pl = &sXL[mrow * WP + kA];
            union { v16h h; v4f f[2]; } ah, al;
            ah.f[0] = *(const v4f*)ph;  ah.f[1] = *(const v4f*)(ph + 16);
            al.f[0] = *(const v4f*)pl;  al.f[1] = *(const v4f*)(pl + 16);
            #pragma unroll
            for (int ni = 0; ni < 4; ++ni) {
                acc[mi][ni] = __builtin_amdgcn_wmma_f32_16x16x32_f16(
                    false, ah.h, false, bh[ni], (short)0, acc[mi][ni], false, false);
                acc[mi][ni] = __builtin_amdgcn_wmma_f32_16x16x32_f16(
                    false, al.h, false, bh[ni], (short)0, acc[mi][ni], false, false);
                acc[mi][ni] = __builtin_amdgcn_wmma_f32_16x16x32_f16(
                    false, ah.h, false, bl[ni], (short)0, acc[mi][ni], false, false);
            }
        }
        __syncthreads();
    }

    // epilogue: C layout = VGPR r -> row (lane<16 ? r : r+8), col = lane&15
    #pragma unroll
    for (int ni = 0; ni < 4; ++ni) {
        int col = n0 + wn * 64 + ni * 16 + l16;
        float bv = bias[col];
        #pragma unroll
        for (int mi = 0; mi < 4; ++mi) {
            int mbase = m0 + wm * 64 + mi * 16 + ((lane >> 4) << 3);
            #pragma unroll
            for (int r = 0; r < 8; ++r)
                out[(size_t)(mbase + r) * OUT_F + col] = acc[mi][ni][r] + bv;
        }
    }
}

// ------------------------------------------------------------------
extern "C" void kernel_launch(void* const* d_in, const int* in_sizes, int n_in,
                              void* d_out, int out_size, void* d_ws, size_t ws_size,
                              hipStream_t stream) {
    const float* x       = (const float*)d_in[0];
    const float* w0      = (const float*)d_in[1];
    const float* lora_a  = (const float*)d_in[2];
    const float* lora_b  = (const float*)d_in[3];
    const float* q_scale = (const float*)d_in[4];
    const float* bias    = (const float*)d_in[5];
    float*       outp    = (float*)d_out;

    const int    M  = out_size / OUT_F;                    // 8192
    const size_t NW = (size_t)OUT_F * IN_F;                // 16.78M
    const size_t NX = (size_t)M * IN_F;                    // 33.55M

    _Float16* whi = (_Float16*)d_ws;                       // 32 MB
    _Float16* wlo = whi + NW;                              // 32 MB
    _Float16* xhi = wlo + NW;                              // 64 MB
    _Float16* xlo = xhi + NX;                              // 64 MB

    dim3 g1(IN_F / BI, OUT_F / BO);                        // (16, 64)
    wq_split_kernel<<<g1, 256, 0, stream>>>(w0, lora_a, lora_b, q_scale, whi, wlo);

    x_split_kernel<<<(unsigned)(NX / 4 / 256), 256, 0, stream>>>(x, xhi, xlo);

    dim3 g2(OUT_F / BN, M / BM);                           // (16, 64)
    gemm_wmma_kernel<<<g2, 256, 0, stream>>>(xhi, xlo, whi, wlo, bias, outp);
}
